// ProxyInitializer_34557306864137
// MI455X (gfx1250) — matrix operand order repaired and checked by reading, not compiled, not run on hardware
//
#include <hip/hip_runtime.h>

typedef __attribute__((ext_vector_type(16))) __bf16 v16bf;
typedef __attribute__((ext_vector_type(8)))  float  v8f;
typedef __attribute__((ext_vector_type(2)))  __bf16 bf2;
typedef __attribute__((address_space(3)))    float  lds_f32;

#define DH    4096   // hidden dim D
#define NTOK  2304   // N tokens
#define NPX   256    // proxy tokens
#define BATCH 8
#define SPK   68     // padded LDS stride for [k][n] f32 tiles (floats)
#define SPN   36     // padded LDS stride for [n][k] f32 tiles (floats)

__device__ __forceinline__ v8f wmma_bf16(v16bf a, v16bf b, v8f c) {
  return __builtin_amdgcn_wmma_f32_16x16x32_bf16(
      /*neg_a=*/false, a, /*neg_b=*/false, b,
      /*c_mod=*/(short)0, c, /*reuse_a=*/false, /*reuse_b=*/false);
}

__device__ __forceinline__ unsigned lds_addr(float* p) {
  // generic (flat) -> LDS address space -> 32-bit LDS byte offset
  return (unsigned)(unsigned long long)(lds_f32*)p;
}

__device__ __forceinline__ void async_b128(unsigned lds, const float* g) {
  asm volatile("global_load_async_to_lds_b128 %0, %1, off"
               :: "v"(lds), "v"((unsigned long long)g)
               : "memory");
}

// ---- stage 32(k) x 64(n) f32 tile, source B[k][n] = src[k*DH + n] ---------
// LDS layout [k][n], padded row stride SPK. 128 threads, 4 async b128 each.
__device__ __forceinline__ void stage_kmajor(const float* src, float* buf) {
  const int tid = threadIdx.x;
  #pragma unroll
  for (int c = 0; c < 4; ++c) {
    const int cid = tid + 128 * c;        // 0..511 chunks of 4 floats
    const int k   = cid >> 4;             // 0..31
    const int n4  = (cid & 15) << 2;      // 0,4,..,60
    async_b128(lds_addr(buf + k * SPK + n4), src + (size_t)k * DH + n4);
  }
}

// ---- stage 32(k) x 64(n) f32 tile, source B[k][n] = src[n*DH + k] ---------
// LDS layout [n][k], padded row stride SPN. Per-thread source is k-contiguous.
__device__ __forceinline__ void stage_nmajor(const float* src, float* buf) {
  const int n  = threadIdx.x & 63;
  const int kh = (threadIdx.x >> 6) << 4; // 0 or 16
  const float* g = src + (size_t)n * DH + kh;
  float* l = buf + n * SPN + kh;
  #pragma unroll
  for (int c = 0; c < 4; ++c) async_b128(lds_addr(l + 4 * c), g + 4 * c);
}

// ---- A fragments (16x32, lanes0-15 K=0-15, lanes16-31 K=16-31) -----------
__device__ __forceinline__ v16bf afrag_f32(const float* arow, int hi) {
  v16bf a;
  #pragma unroll
  for (int j = 0; j < 8; ++j) {
    const int k0 = 2 * j + hi * 16;
    float2 f = *(const float2*)(arow + k0);
    a[2 * j]     = (__bf16)f.x;
    a[2 * j + 1] = (__bf16)f.y;
  }
  return a;
}
__device__ __forceinline__ v16bf afrag_bf16(const __bf16* arow, int hi) {
  v16bf a;
  #pragma unroll
  for (int j = 0; j < 8; ++j) {
    const int k0 = 2 * j + hi * 16;
    bf2 p = *(const bf2*)(arow + k0);
    a[2 * j]     = p.x;
    a[2 * j + 1] = p.y;
  }
  return a;
}

// ---- B fragments (32x16, lane = column, lanes16-31 hold K=16-31) ----------
__device__ __forceinline__ v16bf bfrag_kmajor(const float* bt, int col, int hi) {
  v16bf bb;
  #pragma unroll
  for (int j = 0; j < 8; ++j) {
    const int k0 = 2 * j + hi * 16;
    float x = bt[k0 * SPK + col];
    float y = bt[(k0 + 1) * SPK + col];
    bb[2 * j]     = (__bf16)x;
    bb[2 * j + 1] = (__bf16)y;
  }
  return bb;
}
__device__ __forceinline__ v16bf bfrag_nmajor(const float* bt, int nrow, int hi) {
  v16bf bb;
  #pragma unroll
  for (int jj = 0; jj < 4; ++jj) {          // 4 x ds_load_b128
    float4 f = *(const float4*)(bt + nrow * SPN + hi * 16 + 4 * jj);
    bb[4 * jj]     = (__bf16)f.x;
    bb[4 * jj + 1] = (__bf16)f.y;
    bb[4 * jj + 2] = (__bf16)f.z;
    bb[4 * jj + 3] = (__bf16)f.w;
  }
  return bb;
}

// ---------------- Kernel 1: P = proxy(256x4096) @ Wk(4096x4096) -> bf16 ----
__global__ __launch_bounds__(128) void k_p_gemm(const float* __restrict__ proxy,
                                                const float* __restrict__ Wk,
                                                __bf16* __restrict__ Pbf) {
  const int lane = threadIdx.x & 31;
  const int wave = threadIdx.x >> 5;
  const int hi = lane >> 4, l16 = lane & 15;
  const int nbase = blockIdx.x * 64;
  const int qrow0 = blockIdx.y * 64 + wave * 16;
  __shared__ float tb[2][32 * SPK];

  v8f acc[4];
  #pragma unroll
  for (int t = 0; t < 4; ++t) acc[t] = (v8f){0.f,0.f,0.f,0.f,0.f,0.f,0.f,0.f};
  const float* arow = proxy + (size_t)(qrow0 + l16) * DH;

  stage_kmajor(Wk + nbase, tb[0]);
  const int NK = DH / 32;
  for (int i = 0; i < NK; ++i) {
    const float* bt = tb[i & 1];
    if (i + 1 < NK) {
      __syncthreads();                                   // done reading tb[(i+1)&1]
      stage_kmajor(Wk + (size_t)(i + 1) * 32 * DH + nbase, tb[(i + 1) & 1]);
      asm volatile("s_wait_asynccnt 0x4" ::: "memory");  // tile i landed
    } else {
      asm volatile("s_wait_asynccnt 0x0" ::: "memory");
    }
    __syncthreads();
    v16bf a = afrag_f32(arow + i * 32, hi);
    #pragma unroll
    for (int t = 0; t < 4; ++t)
      acc[t] = wmma_bf16(a, bfrag_kmajor(bt, t * 16 + l16, hi), acc[t]);
  }
  #pragma unroll
  for (int t = 0; t < 4; ++t)
    #pragma unroll
    for (int j = 0; j < 8; ++j)
      Pbf[(size_t)(qrow0 + j + hi * 8) * DH + nbase + t * 16 + l16] = (__bf16)acc[t][j];
}

// ---------------- Kernel 2: logits[b] = P @ full[b]^T / 64 -----------------
__global__ __launch_bounds__(128) void k_logits(const float* __restrict__ full,
                                                const __bf16* __restrict__ Pbf,
                                                float* __restrict__ logits) {
  const int lane = threadIdx.x & 31;
  const int wave = threadIdx.x >> 5;
  const int hi = lane >> 4, l16 = lane & 15;
  const int nbase = blockIdx.x * 64;
  const int qrow0 = blockIdx.y * 64 + wave * 16;
  const int b = blockIdx.z;
  const float* fb = full + (size_t)b * NTOK * DH;
  const __bf16* arow = Pbf + (size_t)(qrow0 + l16) * DH;
  __shared__ float tb[2][64 * SPN];

  v8f acc[4];
  #pragma unroll
  for (int t = 0; t < 4; ++t) acc[t] = (v8f){0.f,0.f,0.f,0.f,0.f,0.f,0.f,0.f};

  stage_nmajor(fb + (size_t)nbase * DH, tb[0]);
  const int NK = DH / 32;
  for (int i = 0; i < NK; ++i) {
    const float* bt = tb[i & 1];
    if (i + 1 < NK) {
      __syncthreads();
      stage_nmajor(fb + (size_t)nbase * DH + (i + 1) * 32, tb[(i + 1) & 1]);
      asm volatile("s_wait_asynccnt 0x4" ::: "memory");
    } else {
      asm volatile("s_wait_asynccnt 0x0" ::: "memory");
    }
    __syncthreads();
    v16bf a = afrag_bf16(arow + i * 32, hi);
    #pragma unroll
    for (int t = 0; t < 4; ++t)
      acc[t] = wmma_bf16(a, bfrag_nmajor(bt, t * 16 + l16, hi), acc[t]);
  }
  const float scale = 0.015625f;  // 1/sqrt(4096)
  #pragma unroll
  for (int t = 0; t < 4; ++t)
    #pragma unroll
    for (int j = 0; j < 8; ++j)
      logits[((size_t)b * NPX + qrow0 + j + hi * 8) * NTOK + nbase + t * 16 + l16] =
          acc[t][j] * scale;
}

// ---------------- block reductions (blockDim = 256 = 8 waves) --------------
__device__ __forceinline__ float block_reduce_max(float x) {
  __shared__ float smx[8];
  #pragma unroll
  for (int o = 16; o > 0; o >>= 1) x = fmaxf(x, __shfl_xor(x, o, 32));
  if ((threadIdx.x & 31) == 0) smx[threadIdx.x >> 5] = x;
  __syncthreads();
  float r = smx[0];
  #pragma unroll
  for (int i = 1; i < 8; ++i) r = fmaxf(r, smx[i]);
  __syncthreads();
  return r;
}
__device__ __forceinline__ float block_reduce_sum(float x) {
  __shared__ float sms[8];
  #pragma unroll
  for (int o = 16; o > 0; o >>= 1) x += __shfl_xor(x, o, 32);
  if ((threadIdx.x & 31) == 0) sms[threadIdx.x >> 5] = x;
  __syncthreads();
  float r = sms[0];
  #pragma unroll
  for (int i = 1; i < 8; ++i) r += sms[i];
  __syncthreads();
  return r;
}

// ---------------- Kernel 3: softmax over N (rows of 2304) -> bf16 attn -----
__global__ __launch_bounds__(256) void k_softmax_n(const float* __restrict__ logits,
                                                   __bf16* __restrict__ attnbf) {
  const size_t row = blockIdx.x;  // b*NPX + q
  const float* src = logits + row * NTOK;
  float v[9];
  float m = -__builtin_inff();
  #pragma unroll
  for (int i = 0; i < 9; ++i) {
    v[i] = src[threadIdx.x + 256 * i];
    m = fmaxf(m, v[i]);
  }
  m = block_reduce_max(m);
  float s = 0.f;
  #pragma unroll
  for (int i = 0; i < 9; ++i) { v[i] = __expf(v[i] - m); s += v[i]; }
  s = block_reduce_sum(s);
  const float inv = 1.f / s;
  __bf16* dst = attnbf + row * NTOK;
  #pragma unroll
  for (int i = 0; i < 9; ++i) dst[threadIdx.x + 256 * i] = (__bf16)(v[i] * inv);
}

// ---------------- Kernel 4: proxyv_out[b] = attn[b] @ full[b] --------------
__global__ __launch_bounds__(128) void k_out_gemm(const float* __restrict__ full,
                                                  const __bf16* __restrict__ attnbf,
                                                  float* __restrict__ out) {
  const int lane = threadIdx.x & 31;
  const int wave = threadIdx.x >> 5;
  const int hi = lane >> 4, l16 = lane & 15;
  const int dbase = blockIdx.x * 64;
  const int qrow0 = blockIdx.y * 64 + wave * 16;
  const int b = blockIdx.z;
  const float* fb = full + (size_t)b * NTOK * DH;
  const __bf16* arow = attnbf + ((size_t)b * NPX + qrow0 + l16) * NTOK;
  __shared__ float tb[2][32 * SPK];

  v8f acc[4];
  #pragma unroll
  for (int t = 0; t < 4; ++t) acc[t] = (v8f){0.f,0.f,0.f,0.f,0.f,0.f,0.f,0.f};

  stage_kmajor(fb + dbase, tb[0]);
  const int NK = NTOK / 32;  // 72
  for (int i = 0; i < NK; ++i) {
    const float* bt = tb[i & 1];
    if (i + 1 < NK) {
      __syncthreads();
      stage_kmajor(fb + (size_t)(i + 1) * 32 * DH + dbase, tb[(i + 1) & 1]);
      asm volatile("s_wait_asynccnt 0x4" ::: "memory");
    } else {
      asm volatile("s_wait_asynccnt 0x0" ::: "memory");
    }
    __syncthreads();
    v16bf a = afrag_bf16(arow + i * 32, hi);
    #pragma unroll
    for (int t = 0; t < 4; ++t)
      acc[t] = wmma_bf16(a, bfrag_kmajor(bt, t * 16 + l16, hi), acc[t]);
  }
  #pragma unroll
  for (int t = 0; t < 4; ++t)
    #pragma unroll
    for (int j = 0; j < 8; ++j)
      out[((size_t)b * NPX + qrow0 + j + hi * 8) * DH + dbase + t * 16 + l16] = acc[t][j];
}

// ---------------- Kernel 5: softmax over q (256) -> attn_T (B,N,256) -------
__global__ __launch_bounds__(256) void k_softmax_q(const float* __restrict__ logits,
                                                   float* __restrict__ attnT) {
  const int n = blockIdx.x;
  const int b = blockIdx.y;
  const int q = threadIdx.x;
  float v = logits[((size_t)b * NPX + q) * NTOK + n];
  float m = block_reduce_max(v);
  float e = __expf(v - m);
  float s = block_reduce_sum(e);
  attnT[((size_t)b * NTOK + n) * NPX + q] = e / s;
}

extern "C" void kernel_launch(void* const* d_in, const int* in_sizes, int n_in,
                              void* d_out, int out_size, void* d_ws, size_t ws_size,
                              hipStream_t stream) {
  (void)in_sizes; (void)n_in; (void)out_size; (void)ws_size;
  const float* full  = (const float*)d_in[0];  // (B, N, D) f32
  const float* Wk    = (const float*)d_in[1];  // (D, D) f32, [out,in]
  const float* proxy = (const float*)d_in[2];  // (n, D) f32

  float* proxyv_out = (float*)d_out;                           // (B, n, D)
  float* attnT      = proxyv_out + (size_t)BATCH * NPX * DH;   // (B, N, n)

  char* ws = (char*)d_ws;
  __bf16* Pbf    = (__bf16*)ws;                                          // 2 MB
  float*  logits = (float*)(ws + (size_t)NPX * DH * 2);                  // 18.9 MB
  __bf16* attnbf = (__bf16*)(ws + (size_t)NPX * DH * 2 +
                             (size_t)BATCH * NPX * NTOK * 4);            // 9.4 MB

  k_p_gemm  <<<dim3(DH / 64, NPX / 64),          dim3(128), 0, stream>>>(proxy, Wk, Pbf);
  k_logits  <<<dim3(NTOK / 64, NPX / 64, BATCH), dim3(128), 0, stream>>>(full, Pbf, logits);
  k_softmax_n<<<dim3(BATCH * NPX),               dim3(256), 0, stream>>>(logits, attnbf);
  k_out_gemm<<<dim3(DH / 64, NPX / 64, BATCH),   dim3(128), 0, stream>>>(full, attnbf, proxyv_out);
  k_softmax_q<<<dim3(NTOK, BATCH),               dim3(256), 0, stream>>>(logits, attnT);
}
